// LinearAttention_69535520522778
// MI455X (gfx1250) — matrix operand order repaired and checked by reading, not compiled, not run on hardware
//
#include <hip/hip_runtime.h>
#include <hip/hip_bf16.h>

#define T_DIM  4096
#define IN_DIM 512
#define K_DIM  64
#define V_DIM  64
#define H_DIM  4096
#define NCHUNK 64
#define CHUNK  64
#define LDST   40   // padded LDS row stride (bf16 elems): 80B = 20 banks, 16B-aligned

typedef __bf16 bf16_t;
typedef bf16_t bf16x16 __attribute__((ext_vector_type(16)));
typedef bf16_t bf16x8  __attribute__((ext_vector_type(8)));
typedef float  f32x8   __attribute__((ext_vector_type(8)));

__device__ __forceinline__ float phi_f(float x) {           // 1 + elu(x)
    return x > 0.f ? x + 1.f : expf(x);
}
__device__ __forceinline__ float mish_f(float x) {
    float sp = (x > 20.f) ? x : log1pf(expf(x));
    return x * tanhf(sp);
}

// ---------------------------------------------------------------------------
// K0: k = phi(x Wk^T), q = phi(x Wq^T), v = x Wv^T + bv
// ---------------------------------------------------------------------------
__global__ void __launch_bounds__(256)
proj_kernel(const float* __restrict__ x,
            const float* __restrict__ Wk, const float* __restrict__ Wq,
            const float* __restrict__ Wv, const float* __restrict__ bv,
            float* __restrict__ ko, float* __restrict__ qo, float* __restrict__ vo)
{
    __shared__ float xs[IN_DIM];
    const int t = blockIdx.x, tid = threadIdx.x;
    for (int i = tid; i < IN_DIM; i += 256) xs[i] = x[t * IN_DIM + i];
    __syncthreads();
    if (tid < 192) {
        const int which = tid >> 6;       // 0:k 1:q 2:v
        const int j = tid & 63;
        const float* W = (which == 0) ? Wk : (which == 1) ? Wq : Wv;
        const float* row = W + j * IN_DIM;
        float acc = 0.f;
#pragma unroll 8
        for (int i = 0; i < IN_DIM; ++i) acc = fmaf(xs[i], row[i], acc);
        if (which == 0)      ko[t * K_DIM + j] = phi_f(acc);
        else if (which == 1) qo[t * K_DIM + j] = phi_f(acc);
        else                 vo[t * V_DIM + j] = acc + bv[j];
    }
}

// ---------------------------------------------------------------------------
// K1: per-chunk totals of the reset-scan (state kept in registers)
// ---------------------------------------------------------------------------
__global__ void __launch_bounds__(256)
scan_chunk_kernel(const float* __restrict__ k, const float* __restrict__ v,
                  const int* __restrict__ start,
                  float* __restrict__ ssum, float* __restrict__ zsum,
                  int* __restrict__ rflag)
{
    __shared__ float kr[CHUNK * K_DIM];
    __shared__ float vr[CHUNK * V_DIM];
    __shared__ int   fl[CHUNK];
    const int tid = threadIdx.x, c = blockIdx.x;
#pragma unroll
    for (int e = 0; e < 16; ++e) {
        int idx = tid + 256 * e;
        kr[idx] = k[c * CHUNK * K_DIM + idx];
        vr[idx] = v[c * CHUNK * V_DIM + idx];
    }
    if (tid < CHUNK) fl[tid] = start[c * CHUNK + tid];
    __syncthreads();

    float acc[16];
#pragma unroll
    for (int e = 0; e < 16; ++e) acc[e] = 0.f;
    float zacc = 0.f;
    const int j = tid & 63, ib = tid >> 6;
    int any = 0;
    for (int t = 0; t < CHUNK; ++t) {
        const int f = fl[t];
        any |= f;
        const float vv = vr[t * V_DIM + j];
#pragma unroll
        for (int e = 0; e < 16; ++e) {
            float kk = kr[t * K_DIM + ib + 4 * e];
            acc[e] = f ? (kk * vv) : fmaf(kk, vv, acc[e]);
        }
        if (tid < K_DIM) {
            float kz = kr[t * K_DIM + tid];
            zacc = f ? kz : (zacc + kz);
        }
    }
#pragma unroll
    for (int e = 0; e < 16; ++e) ssum[c * H_DIM + tid + 256 * e] = acc[e];
    if (tid < K_DIM) zsum[c * K_DIM + tid] = zacc;
    if (tid == 0)    rflag[c] = any;
}

// ---------------------------------------------------------------------------
// K2: sequential scan over the 64 chunk summaries -> exclusive carries
// ---------------------------------------------------------------------------
__global__ void __launch_bounds__(256)
scan_carry_kernel(const float* __restrict__ s0, const float* __restrict__ z0,
                  const float* __restrict__ ssum, const float* __restrict__ zsum,
                  const int* __restrict__ rflag,
                  float* __restrict__ carryS, float* __restrict__ carryZ)
{
    const int tid = threadIdx.x;
    float cs[16];
#pragma unroll
    for (int e = 0; e < 16; ++e) cs[e] = s0[tid + 256 * e];
    float cz = (tid < K_DIM) ? z0[tid] : 0.f;
    for (int c = 0; c < NCHUNK; ++c) {
#pragma unroll
        for (int e = 0; e < 16; ++e) carryS[c * H_DIM + tid + 256 * e] = cs[e];
        if (tid < K_DIM) carryZ[c * K_DIM + tid] = cz;
        const int f = rflag[c];
#pragma unroll
        for (int e = 0; e < 16; ++e)
            cs[e] = (f ? 0.f : cs[e]) + ssum[c * H_DIM + tid + 256 * e];
        if (tid < K_DIM) cz = (f ? 0.f : cz) + zsum[c * K_DIM + tid];
    }
}

// ---------------------------------------------------------------------------
// K3: replay each chunk with its carry; write s,z directly to d_out regions
// ---------------------------------------------------------------------------
__global__ void __launch_bounds__(256)
scan_final_kernel(const float* __restrict__ k, const float* __restrict__ v,
                  const int* __restrict__ start,
                  const float* __restrict__ carryS, const float* __restrict__ carryZ,
                  float* __restrict__ sOut, float* __restrict__ zOut)
{
    __shared__ float kr[CHUNK * K_DIM];
    __shared__ float vr[CHUNK * V_DIM];
    __shared__ int   fl[CHUNK];
    const int tid = threadIdx.x, c = blockIdx.x;
#pragma unroll
    for (int e = 0; e < 16; ++e) {
        int idx = tid + 256 * e;
        kr[idx] = k[c * CHUNK * K_DIM + idx];
        vr[idx] = v[c * CHUNK * V_DIM + idx];
    }
    if (tid < CHUNK) fl[tid] = start[c * CHUNK + tid];
    __syncthreads();

    float st[16];
#pragma unroll
    for (int e = 0; e < 16; ++e) st[e] = carryS[c * H_DIM + tid + 256 * e];
    float zst = (tid < K_DIM) ? carryZ[c * K_DIM + tid] : 0.f;
    const int j = tid & 63, ib = tid >> 6;
    for (int t = 0; t < CHUNK; ++t) {
        const int f = fl[t];
        const float vv = vr[t * V_DIM + j];
        const int trow = c * CHUNK + t;
#pragma unroll
        for (int e = 0; e < 16; ++e) {
            float kk = kr[t * K_DIM + ib + 4 * e];
            st[e] = f ? (kk * vv) : fmaf(kk, vv, st[e]);
            sOut[(size_t)trow * H_DIM + tid + 256 * e] = st[e];
        }
        if (tid < K_DIM) {
            float kz = kr[t * K_DIM + tid];
            zst = f ? kz : (zst + kz);
            zOut[trow * K_DIM + tid] = zst;
        }
    }
}

// ---------------------------------------------------------------------------
// K4: out[t,j] = (sum_i s[t,i,j] q[t,i]) / max((sum z)*(sum q), 1e-6)
// ---------------------------------------------------------------------------
__global__ void __launch_bounds__(64)
attn_out_kernel(const float* __restrict__ q, const float* __restrict__ sD,
                const float* __restrict__ zD, float* __restrict__ out)
{
    const int t = blockIdx.x, j = threadIdx.x;
    __shared__ float qs[K_DIM];
    __shared__ float zs[K_DIM];
    qs[j] = q[t * K_DIM + j];
    zs[j] = zD[t * K_DIM + j];
    __syncthreads();
    float sq = 0.f, sz = 0.f;
#pragma unroll
    for (int i = 0; i < K_DIM; ++i) { sq += qs[i]; sz += zs[i]; }
    const float denom = fmaxf(sz * sq, 1e-6f);
    float num = 0.f;
#pragma unroll 8
    for (int i = 0; i < K_DIM; ++i)
        num = fmaf(sD[(size_t)t * H_DIM + i * V_DIM + j], qs[i], num);
    out[t * V_DIM + j] = num / denom;
}

// ---------------------------------------------------------------------------
// Fragment loaders (both A and B tiles stored row-major [row][k], stride LDST)
//  A frag (ISA 16-bit A 16x32): lane row = m, elems = [8g..8g+7] ++ [16+8g..16+8g+7]
//  B frag (ISA 16-bit B 32x16): lane row = n, elems = [8g..8g+7] ++ [16+8g..16+8g+7]
// ---------------------------------------------------------------------------
__device__ __forceinline__ bf16x16 cat8(bf16x8 a, bf16x8 b) {
    return __builtin_shufflevector(a, b, 0,1,2,3,4,5,6,7,8,9,10,11,12,13,14,15);
}
__device__ __forceinline__ bf16x16 frag_a(const bf16_t* __restrict__ p, int g8) {
    bf16x8 lo = *(const bf16x8*)(p + g8);
    bf16x8 hi = *(const bf16x8*)(p + 16 + g8);
    return cat8(lo, hi);
}
__device__ __forceinline__ bf16x16 frag_b(const bf16_t* __restrict__ p, int g16) {
    bf16x8 lo = *(const bf16x8*)(p + g16);
    bf16x8 hi = *(const bf16x8*)(p + 16 + g16);
    return cat8(lo, hi);
}

// ---------------------------------------------------------------------------
// bf16x3 WMMA GEMM: C[m,n] = act( sum_k A1[m,k] B1[n,k] (+ A2 B2) + bias )
// Block: 256 thr (8 waves), tile 128x128; wave tile 64x32 (4x2 wmma tiles).
// Register-prefetch pipeline: next k-tile's global loads issued before WMMAs.
// ---------------------------------------------------------------------------
__global__ void __launch_bounds__(256)
gemm_bf16x3_kernel(const float* __restrict__ A1, const float* __restrict__ B1, int K1,
                   const float* __restrict__ A2, const float* __restrict__ B2, int K2,
                   const float* __restrict__ bias1, const float* __restrict__ bias2,
                   float* __restrict__ C, int N, int act)
{
    __shared__ bf16_t sAh[128 * LDST];
    __shared__ bf16_t sAl[128 * LDST];
    __shared__ bf16_t sBh[128 * LDST];
    __shared__ bf16_t sBl[128 * LDST];

    const int tid = threadIdx.x, lane = tid & 31, wid = tid >> 5;
    const int nb = blockIdx.x * 128, mb = blockIdx.y * 128;
    const int wm = (wid >> 2) * 64, wn = (wid & 3) * 32;
    const int g8 = (lane >> 4) * 8, g16 = (lane >> 4) * 16, lr = lane & 15;

    f32x8 acc[4][2];
#pragma unroll
    for (int mi = 0; mi < 4; ++mi)
#pragma unroll
        for (int ni = 0; ni < 2; ++ni) acc[mi][ni] = (f32x8)0.f;

    for (int seg = 0; seg < 2; ++seg) {
        const float* A = seg ? A2 : A1;
        const float* B = seg ? B2 : B1;
        const int    K = seg ? K2 : K1;
        if (K == 0) continue;

        float aR[16], bR[16];
#pragma unroll
        for (int e = 0; e < 16; ++e) {
            int idx = tid + 256 * e, m = idx >> 5, kk = idx & 31;
            aR[e] = A[(size_t)(mb + m) * K + kk];
            bR[e] = B[(size_t)(nb + m) * K + kk];
        }

        for (int kb = 0; kb < K; kb += 32) {
            __syncthreads();   // previous iteration's frag reads are done
#pragma unroll
            for (int e = 0; e < 16; ++e) {
                int idx = tid + 256 * e, m = idx >> 5, kk = idx & 31;
                bf16_t ah = (bf16_t)aR[e];
                sAh[m * LDST + kk] = ah;
                sAl[m * LDST + kk] = (bf16_t)(aR[e] - (float)ah);
                bf16_t bh = (bf16_t)bR[e];
                sBh[m * LDST + kk] = bh;
                sBl[m * LDST + kk] = (bf16_t)(bR[e] - (float)bh);
            }
            __syncthreads();

            if (kb + 32 < K) {  // prefetch next k-tile; overlaps with WMMAs below
#pragma unroll
                for (int e = 0; e < 16; ++e) {
                    int idx = tid + 256 * e, m = idx >> 5, kk = idx & 31;
                    aR[e] = A[(size_t)(mb + m) * K + kb + 32 + kk];
                    bR[e] = B[(size_t)(nb + m) * K + kb + 32 + kk];
                }
            }

            bf16x16 ah[4], al[4], bh[2], bl[2];
#pragma unroll
            for (int mi = 0; mi < 4; ++mi) {
                int row = (wm + 16 * mi + lr) * LDST;
                ah[mi] = frag_a(sAh + row, g8);
                al[mi] = frag_a(sAl + row, g8);
            }
#pragma unroll
            for (int ni = 0; ni < 2; ++ni) {
                int row = (wn + 16 * ni + lr) * LDST;
                bh[ni] = frag_b(sBh + row, g8);
                bl[ni] = frag_b(sBl + row, g8);
            }
#pragma unroll
            for (int mi = 0; mi < 4; ++mi)
#pragma unroll
                for (int ni = 0; ni < 2; ++ni) {
                    acc[mi][ni] = __builtin_amdgcn_wmma_f32_16x16x32_bf16(
                        false, ah[mi], false, bh[ni], (short)0, acc[mi][ni], false, false);
                    acc[mi][ni] = __builtin_amdgcn_wmma_f32_16x16x32_bf16(
                        false, al[mi], false, bh[ni], (short)0, acc[mi][ni], false, false);
                    acc[mi][ni] = __builtin_amdgcn_wmma_f32_16x16x32_bf16(
                        false, ah[mi], false, bl[ni], (short)0, acc[mi][ni], false, false);
                }
        }
    }

    // epilogue: C/D 16x16 f32 layout (VGPR r -> row r + 8*(lane/16), col lane%16)
#pragma unroll
    for (int mi = 0; mi < 4; ++mi)
#pragma unroll
        for (int ni = 0; ni < 2; ++ni)
#pragma unroll
            for (int r = 0; r < 8; ++r) {
                int m = mb + wm + 16 * mi + r + 8 * (lane >> 4);
                int n = nb + wn + 16 * ni + (lane & 15);
                float v = acc[mi][ni][r] + bias1[n] + (bias2 ? bias2[n] : 0.f);
                if (act) v = mish_f(v);
                C[(size_t)m * N + n] = v;
            }
}

// ---------------------------------------------------------------------------
// K8: layernorm over H (two-pass to match reference var = mean((h-mu)^2))
// ---------------------------------------------------------------------------
__global__ void __launch_bounds__(256)
layernorm_kernel(const float* __restrict__ h, const float* __restrict__ g,
                 const float* __restrict__ b, float* __restrict__ outp)
{
    const int t = blockIdx.x, tid = threadIdx.x;
    __shared__ float red[256];
    const float* row = h + (size_t)t * H_DIM;

    float s = 0.f;
    for (int i = tid; i < H_DIM; i += 256) s += row[i];
    red[tid] = s; __syncthreads();
    for (int off = 128; off > 0; off >>= 1) {
        if (tid < off) red[tid] += red[tid + off];
        __syncthreads();
    }
    const float mean = red[0] / (float)H_DIM;
    __syncthreads();

    float s2 = 0.f;
    for (int i = tid; i < H_DIM; i += 256) { float d = row[i] - mean; s2 = fmaf(d, d, s2); }
    red[tid] = s2; __syncthreads();
    for (int off = 128; off > 0; off >>= 1) {
        if (tid < off) red[tid] += red[tid + off];
        __syncthreads();
    }
    const float inv = rsqrtf(red[0] / (float)H_DIM + 1e-5f);
    for (int i = tid; i < H_DIM; i += 256)
        outp[(size_t)t * H_DIM + i] = (row[i] - mean) * inv * g[i] + b[i];
}

// ---------------------------------------------------------------------------
extern "C" void kernel_launch(void* const* d_in, const int* in_sizes, int n_in,
                              void* d_out, int out_size, void* d_ws, size_t ws_size,
                              hipStream_t stream)
{
    const float* x     = (const float*)d_in[0];
    const float* s0    = (const float*)d_in[1];
    const float* z0    = (const float*)d_in[2];
    const int*   start = (const int*)d_in[3];
    // d_in[4] = next_done (unused by the math)
    const float* Wk    = (const float*)d_in[5];
    const float* Wq    = (const float*)d_in[6];
    const float* Wv    = (const float*)d_in[7];
    const float* bv    = (const float*)d_in[8];
    const float* Wskip = (const float*)d_in[9];
    const float* bskip = (const float*)d_in[10];
    const float* W1    = (const float*)d_in[11];
    const float* b1    = (const float*)d_in[12];
    const float* W2    = (const float*)d_in[13];
    const float* b2    = (const float*)d_in[14];
    const float* W3    = (const float*)d_in[15];
    const float* b3    = (const float*)d_in[16];
    const float* ln_g  = (const float*)d_in[17];
    const float* ln_b  = (const float*)d_in[18];

    float* outp = (float*)d_out;
    float* hn   = outp;                                  // T*H
    float* sOut = outp + (size_t)T_DIM * H_DIM;          // T*K*V
    float* zOut = sOut + (size_t)T_DIM * K_DIM * V_DIM;  // T*K

    float* ws     = (float*)d_ws;
    float* kbuf   = ws;                                  // T*K
    float* qbuf   = kbuf + T_DIM * K_DIM;                // T*K
    float* vbuf   = qbuf + T_DIM * K_DIM;                // T*V
    float* ssum   = vbuf + T_DIM * V_DIM;                // NCHUNK*H
    float* zsum   = ssum + NCHUNK * H_DIM;               // NCHUNK*K
    int*   rflag  = (int*)(zsum + NCHUNK * K_DIM);       // NCHUNK
    float* carryS = (float*)(rflag + NCHUNK);            // NCHUNK*H
    float* carryZ = carryS + NCHUNK * H_DIM;             // NCHUNK*K
    float* outb   = carryZ + NCHUNK * K_DIM;             // T*V
    float* hbuf   = outb + T_DIM * V_DIM;                // T*H (h1, later h3)

    proj_kernel<<<T_DIM, 256, 0, stream>>>(x, Wk, Wq, Wv, bv, kbuf, qbuf, vbuf);
    scan_chunk_kernel<<<NCHUNK, 256, 0, stream>>>(kbuf, vbuf, start, ssum, zsum, rflag);
    scan_carry_kernel<<<1, 256, 0, stream>>>(s0, z0, ssum, zsum, rflag, carryS, carryZ);
    scan_final_kernel<<<NCHUNK, 256, 0, stream>>>(kbuf, vbuf, start, carryS, carryZ, sOut, zOut);
    attn_out_kernel<<<T_DIM, 64, 0, stream>>>(qbuf, sOut, zOut, outb);

    dim3 gg(H_DIM / 128, T_DIM / 128);
    // h1 = mish(out @ W1^T + b1), K=64
    gemm_bf16x3_kernel<<<gg, 256, 0, stream>>>(outb, W1, V_DIM,
                                               nullptr, nullptr, 0,
                                               b1, nullptr, hbuf, H_DIM, 1);
    // h2 = mish(h1 @ W2^T + b2) -> parked in hn region
    gemm_bf16x3_kernel<<<gg, 256, 0, stream>>>(hbuf, W2, H_DIM,
                                               nullptr, nullptr, 0,
                                               b2, nullptr, hn, H_DIM, 1);
    // h3 = h2 @ W3^T + b3 + x @ Wskip^T + bskip (skip fused as second K-segment)
    gemm_bf16x3_kernel<<<gg, 256, 0, stream>>>(hn, W3, H_DIM,
                                               x, Wskip, IN_DIM,
                                               b3, bskip, hbuf, H_DIM, 0);
    layernorm_kernel<<<T_DIM, 256, 0, stream>>>(hbuf, ln_g, ln_b, hn);
}
